// SigLIPAttention_70626442215659
// MI455X (gfx1250) — matrix operand order repaired and checked by reading, not compiled, not run on hardware
//
#include <hip/hip_runtime.h>
#include <hip/hip_bf16.h>

// ---------------------------------------------------------------------------
// SigLIP attention for MI455X (gfx1250): bf16 WMMA + TDM async tensor loads.
// ---------------------------------------------------------------------------

typedef __attribute__((ext_vector_type(16))) __bf16 v16bf;
typedef __attribute__((ext_vector_type(8)))  __bf16 v8bf;
typedef __attribute__((ext_vector_type(8)))  float  v8f;
typedef __attribute__((ext_vector_type(4)))  unsigned int v4u;
typedef __attribute__((ext_vector_type(8)))  int    v8i;
typedef __attribute__((ext_vector_type(4)))  int    v4i;

#define H_MODEL 1152
#define N_HEADS 16
#define HEAD_DIM 72
#define BATCH 16
#define SEQ 729
#define ROWS (BATCH * SEQ)          // 11664
#define NPAD 736                    // 23 * 32 key blocks
#define DPAD 96                     // 3 * 32 head-dim k-steps
#define QKV_ELEMS ((size_t)BATCH * N_HEADS * NPAD * DPAD)

#if __has_builtin(__builtin_amdgcn_tensor_load_to_lds) && \
    __has_builtin(__builtin_amdgcn_s_wait_tensorcnt)
#define USE_TDM 1
#else
#define USE_TDM 0
#endif

__device__ __forceinline__ v8f wmma_bf16(v16bf a, v16bf b, v8f c) {
    return __builtin_amdgcn_wmma_f32_16x16x32_bf16(
        /*neg_a=*/false, a, /*neg_b=*/false, b,
        /*c_mod=*/(short)0, c, /*reuse_a=*/false, /*reuse_b=*/false);
}

#if USE_TDM
// Issue a TDM load of a 2D bf16 tile [tile_d1 rows x tile_d0 elems] from
// global (row stride = stride_d0 elems) into LDS at byte offset lds_addr.
// D# layout per CDNA5 ISA ch.8: group0 = control/addresses, group1 = dims.
// This toolchain exposes the 6-arg builtin:
//   (uint32x4 g0, int32x8 g1, int32x4 g2, int32x4 g3, int32x8 gx, i32 cpol)
__device__ __forceinline__ void
tdm_load_tile_bf16(unsigned lds_addr, const void* gaddr,
                   unsigned tile_d0, unsigned tile_d1,
                   unsigned tensor_d0, unsigned tensor_d1,
                   unsigned stride_d0) {
    const unsigned long long ga = (unsigned long long)(uintptr_t)gaddr;
    v4u g0;
    g0.x = 1u;                                        // count=1, user D#
    g0.y = lds_addr;                                  // lds_addr [63:32]
    g0.z = (unsigned)(ga & 0xFFFFFFFFu);              // global_addr[31:0]
    g0.w = (unsigned)((ga >> 32) & 0x01FFFFFFu)       // global_addr[56:32]
         | (2u << 30);                                // type=2 ("image")
    v8i g1;
    g1[0] = (int)(1u << 16);                          // data_size=1 (2B), mask=0
    g1[1] = (int)((tensor_d0 & 0xFFFFu) << 16);       // tensor_dim0[15:0]
    g1[2] = (int)((tensor_d0 >> 16) | ((tensor_d1 & 0xFFFFu) << 16));
    g1[3] = (int)((tensor_d1 >> 16) | ((tile_d0 & 0xFFFFu) << 16));
    g1[4] = (int)(tile_d1 & 0xFFFFu);                 // tile_dim1; tile_dim2=0
    g1[5] = (int)stride_d0;                           // tensor_dim0_stride[31:0]
    g1[6] = 0;                                        // stride hi + dim1_stride lo
    g1[7] = 0;                                        // dim1_stride hi (2D: unused)
    const v4i z4 = {0, 0, 0, 0};                      // groups 2/3: dims >2 unused
    const v8i z8 = {0, 0, 0, 0, 0, 0, 0, 0};
    __builtin_amdgcn_tensor_load_to_lds(g0, g1, z4, z4, z8, 0);
}
#endif

// ---------------------------------------------------------------------------
// Zero-fill padded QKV buffers (pad rows/cols must be exact zeros).
// ---------------------------------------------------------------------------
__global__ void zero_ws(uint4* __restrict__ p, long long n16) {
    long long i = (long long)blockIdx.x * blockDim.x + threadIdx.x;
    const long long stride = (long long)gridDim.x * blockDim.x;
    const uint4 z = {0u, 0u, 0u, 0u};
    for (; i < n16; i += stride) p[i] = z;
}

// ---------------------------------------------------------------------------
// Tiled GEMM  Y = X * W^T + bias   via v_wmma_f32_16x16x32_bf16.
//   MODE 0/1/2 : X = f32 activations, scatter bf16 into Q/K/V [B,H,NPAD,DPAD]
//   MODE 3     : X = bf16 attention output, store f32 into d_out
// Block = 256 threads = 8 waves (4 M-waves x 2 N-waves), tile 128x128.
// Ping-pong LDS + register-staged prefetch: one barrier per K-step.
// ---------------------------------------------------------------------------
template <int MODE>
__global__ __launch_bounds__(256) void
gemm_kern(const float* __restrict__ xf, const __bf16* __restrict__ xb,
          const float* __restrict__ W, const float* __restrict__ bias,
          __bf16* __restrict__ dst_qkv, float* __restrict__ dst_f32) {
    __shared__ __bf16 Xs[2][128][32];
    __shared__ __bf16 Ws[2][128][32];

    const int t    = threadIdx.x;
    const int lane = t & 31;
    const int wave = t >> 5;
    const int wm   = wave >> 1;      // 0..3  -> rows wm*32..wm*32+31
    const int wn   = wave & 1;       // 0..1  -> cols wn*64..wn*64+63
    const int row0 = blockIdx.x * 128;
    const int col0 = blockIdx.y * 128;

    // staging chunk geometry: 512 chunks of 8 elems cover a 128x32 tile
    const int xr0 = t >> 2,          xc0 = (t & 3) << 3;
    const int xr1 = (t + 256) >> 2,  xc1 = ((t + 256) & 3) << 3;
    int gx0 = row0 + xr0; if (gx0 > ROWS - 1) gx0 = ROWS - 1;
    int gx1 = row0 + xr1; if (gx1 > ROWS - 1) gx1 = ROWS - 1;

    auto ldx8 = [&](int grow, int col, int k0) -> v8bf {
        v8bf v;
        if (MODE < 3) {
            const float* s = xf + (size_t)grow * H_MODEL + k0 + col;
#pragma unroll
            for (int i = 0; i < 8; ++i) v[i] = (__bf16)s[i];
        } else {
            v = *(const v8bf*)(xb + (size_t)grow * H_MODEL + k0 + col);
        }
        return v;
    };
    auto ldw8 = [&](int r, int col, int k0) -> v8bf {
        const float* s = W + (size_t)(col0 + r) * H_MODEL + k0 + col;
        v8bf v;
#pragma unroll
        for (int i = 0; i < 8; ++i) v[i] = (__bf16)s[i];
        return v;
    };

    v8f acc[2][4] = {};

    // ---- prologue: stage k0 = 0 into LDS buffer 0 ----
    {
        v8bf x0 = ldx8(gx0, xc0, 0), x1 = ldx8(gx1, xc1, 0);
        v8bf w0 = ldw8(xr0, xc0, 0), w1 = ldw8(xr1, xc1, 0);
        *(v8bf*)&Xs[0][xr0][xc0] = x0;  *(v8bf*)&Xs[0][xr1][xc1] = x1;
        *(v8bf*)&Ws[0][xr0][xc0] = w0;  *(v8bf*)&Ws[0][xr1][xc1] = w1;
    }
    __syncthreads();

    int p = 0;
    for (int k0 = 0; k0 < H_MODEL; k0 += 32) {
        const bool has_next = (k0 + 32) < H_MODEL;

        // ---- issue global loads for next K-tile (overlap with compute) ----
        v8bf nx0, nx1, nw0, nw1;
        if (has_next) {
            nx0 = ldx8(gx0, xc0, k0 + 32);  nx1 = ldx8(gx1, xc1, k0 + 32);
            nw0 = ldw8(xr0, xc0, k0 + 32);  nw1 = ldw8(xr1, xc1, k0 + 32);
        }
        if (k0 + 64 < H_MODEL)   // deep prefetch of the W stream into L2/L0
            __builtin_prefetch(W + (size_t)(col0 + xr0) * H_MODEL + k0 + 64, 0, 1);

        // ---- fragments: 2 A rows x 4 B cols, batched loads then WMMA burst ----
        const int ar  = wm * 32 + (lane & 15);
        const int akb = (lane < 16) ? 0 : 16;
        v16bf a0, a1;
        *(v8bf*)&a0       = *(const v8bf*)&Xs[p][ar][akb];
        *((v8bf*)&a0 + 1) = *(const v8bf*)&Xs[p][ar][akb + 8];
        *(v8bf*)&a1       = *(const v8bf*)&Xs[p][ar + 16][akb];
        *((v8bf*)&a1 + 1) = *(const v8bf*)&Xs[p][ar + 16][akb + 8];

        v16bf bfrag[4];
        const int bkb = (lane < 16) ? 0 : 16;
#pragma unroll
        for (int cg = 0; cg < 4; ++cg) {
            const int br = wn * 64 + cg * 16 + (lane & 15);
            *(v8bf*)&bfrag[cg]       = *(const v8bf*)&Ws[p][br][bkb];
            *((v8bf*)&bfrag[cg] + 1) = *(const v8bf*)&Ws[p][br][bkb + 8];
        }
#pragma unroll
        for (int cg = 0; cg < 4; ++cg) {
            acc[0][cg] = wmma_bf16(a0, bfrag[cg], acc[0][cg]);
            acc[1][cg] = wmma_bf16(a1, bfrag[cg], acc[1][cg]);
        }

        // ---- stage next tile into the other buffer; one barrier per step ----
        if (has_next) {
            const int q = p ^ 1;
            *(v8bf*)&Xs[q][xr0][xc0] = nx0;  *(v8bf*)&Xs[q][xr1][xc1] = nx1;
            *(v8bf*)&Ws[q][xr0][xc0] = nw0;  *(v8bf*)&Ws[q][xr1][xc1] = nw1;
            __syncthreads();
            p = q;
        }
    }

    // ---- store: C layout VGPR j -> row j (lanes 0-15) / j+8 (lanes 16-31) ----
#pragma unroll
    for (int sub = 0; sub < 2; ++sub) {
        const int rbase = row0 + wm * 32 + sub * 16 + ((lane < 16) ? 0 : 8);
#pragma unroll
        for (int cg = 0; cg < 4; ++cg) {
            const int o  = col0 + wn * 64 + cg * 16 + (lane & 15);
            const float bb = bias[o];
#pragma unroll
            for (int j = 0; j < 8; ++j) {
                const int m = rbase + j;
                if (m >= ROWS) continue;
                const float val = acc[sub][cg][j] + bb;
                if (MODE < 3) {
                    const int b = m / SEQ, n = m % SEQ;
                    const int h = o / HEAD_DIM, d = o % HEAD_DIM;
                    dst_qkv[((size_t)(b * N_HEADS + h) * NPAD + n) * DPAD + d] =
                        (__bf16)val;
                } else {
                    dst_f32[(size_t)m * H_MODEL + o] = val;
                }
            }
        }
    }
}

// ---------------------------------------------------------------------------
// Flash-style attention: one (b,h) x 64 query rows per 128-thread block.
// Each wave owns 16 query rows; loop over 23 key blocks of 32.
// K block is staged LDS-direct by the Tensor Data Mover (async, TENSORcnt);
// V block is transposed through VGPRs in parallel with the TDM transfer.
// ---------------------------------------------------------------------------
__global__ __launch_bounds__(128) void
attn_wmma(const __bf16* __restrict__ Q, const __bf16* __restrict__ K,
          const __bf16* __restrict__ V, __bf16* __restrict__ O) {
    __shared__ __bf16 Ks[32][DPAD];      // key block, row-major (TDM target)
    __shared__ __bf16 Vt[DPAD][32];      // value block, transposed
    __shared__ __bf16 Pst[4][16][32];    // per-wave P staging

    const int t    = threadIdx.x;
    const int lane = t & 31;
    const int wave = t >> 5;
    const int bh   = blockIdx.y;                 // b*16 + h
    const int qrow0 = blockIdx.x * 64 + wave * 16;
    const float scale = 0.11785113019775792f;    // 72^-0.5

    const __bf16* Qb = Q + (size_t)bh * NPAD * DPAD;
    const __bf16* Kb = K + (size_t)bh * NPAD * DPAD;
    const __bf16* Vb = V + (size_t)bh * NPAD * DPAD;

#if USE_TDM
    const unsigned ks_lds = (unsigned)(uintptr_t)(void*)&Ks[0][0];
#endif

    // ---- load 3 Q A-fragments (K-steps of 32 over DPAD=96) ----
    int qn = qrow0 + (lane & 15);
    if (qn > SEQ - 1) qn = SEQ - 1;              // clamp (store-masked later)
    const __bf16* qrow = Qb + (size_t)qn * DPAD;
    const int akb = (lane < 16) ? 0 : 16;
    v16bf aq[3];
#pragma unroll
    for (int s = 0; s < 3; ++s) {
        *(v8bf*)&aq[s]       = *(const v8bf*)(qrow + s * 32 + akb);
        *((v8bf*)&aq[s] + 1) = *(const v8bf*)(qrow + s * 32 + akb + 8);
    }

    float mrow[8], lrow[8];
#pragma unroll
    for (int j = 0; j < 8; ++j) { mrow[j] = -3.0e38f; lrow[j] = 0.0f; }
    v8f acc[5] = {};                             // d tiles 0..79 (d>=72 masked)

    for (int kb = 0; kb < NPAD / 32; ++kb) {
        __syncthreads();                         // LDS reusable (prev compute done)

#if USE_TDM
        // ---- K block: async DMA straight into LDS (wave 0 issues) ----
        if (wave == 0) {
            tdm_load_tile_bf16(ks_lds, Kb + (size_t)(kb * 32) * DPAD,
                               /*tile_d0=*/DPAD, /*tile_d1=*/32,
                               /*tensor_d0=*/DPAD, /*tensor_d1=*/NPAD,
                               /*stride_d0=*/DPAD);
        }
#endif
        // ---- V block transposed through VGPRs (overlaps the TDM) ----
#pragma unroll
        for (int i = 0; i < 3; ++i) {
            const int chunk = t + i * 128;       // 384 chunks of 8 elems
            const int r  = (chunk * 8) / DPAD;
            const int cc = (chunk * 8) % DPAD;
            const size_t goff = (size_t)(kb * 32 + r) * DPAD + cc;
#if !USE_TDM
            *(v8bf*)&Ks[r][cc] = *(const v8bf*)(Kb + goff);
#endif
            const v8bf vv = *(const v8bf*)(Vb + goff);
#pragma unroll
            for (int e = 0; e < 8; ++e) Vt[cc + e][r] = vv[e];
        }
#if USE_TDM
        if (wave == 0) __builtin_amdgcn_s_wait_tensorcnt(0);
#endif
        __syncthreads();

        // ---- scores: S = Q * K^T  (two 16-key tiles, 3 k-steps) ----
        v8f s0 = {}, s1 = {};
#pragma unroll
        for (int s = 0; s < 3; ++s) {
            const int bkb = s * 32 + ((lane < 16) ? 0 : 16);
            v16bf b0, b1;
            *(v8bf*)&b0       = *(const v8bf*)&Ks[lane & 15][bkb];
            *((v8bf*)&b0 + 1) = *(const v8bf*)&Ks[lane & 15][bkb + 8];
            *(v8bf*)&b1       = *(const v8bf*)&Ks[16 + (lane & 15)][bkb];
            *((v8bf*)&b1 + 1) = *(const v8bf*)&Ks[16 + (lane & 15)][bkb + 8];
            s0 = wmma_bf16(aq[s], b0, s0);
            s1 = wmma_bf16(aq[s], b1, s1);
        }

        // ---- scale + mask padded keys ----
        const int key0 = kb * 32 + (lane & 15);
        const int key1 = key0 + 16;
        const bool ok0 = key0 < SEQ, ok1 = key1 < SEQ;
#pragma unroll
        for (int j = 0; j < 8; ++j) {
            s0[j] = ok0 ? s0[j] * scale : -3.0e38f;
            s1[j] = ok1 ? s1[j] * scale : -3.0e38f;
        }

        // ---- online softmax per row (rows live in 16-lane halves) ----
#pragma unroll
        for (int j = 0; j < 8; ++j) {
            float rm = fmaxf(s0[j], s1[j]);
#pragma unroll
            for (int msk = 1; msk < 16; msk <<= 1)
                rm = fmaxf(rm, __shfl_xor(rm, msk, 32));
            const float mnew = fmaxf(mrow[j], rm);
            const float crs  = __expf(mrow[j] - mnew);
            const float p0   = __expf(s0[j] - mnew);
            const float p1   = __expf(s1[j] - mnew);
            float rs = p0 + p1;
#pragma unroll
            for (int msk = 1; msk < 16; msk <<= 1)
                rs += __shfl_xor(rs, msk, 32);
            lrow[j] = lrow[j] * crs + rs;
            mrow[j] = mnew;
#pragma unroll
            for (int tt = 0; tt < 5; ++tt) acc[tt][j] *= crs;
            // stage P for the A-fragment re-layout
            const int prow = j + ((lane < 16) ? 0 : 8);
            Pst[wave][prow][lane & 15]        = (__bf16)p0;
            Pst[wave][prow][16 + (lane & 15)] = (__bf16)p1;
        }

        // ---- P A-fragment ----
        v16bf pa;
        const int pr = lane & 15;
        *(v8bf*)&pa       = *(const v8bf*)&Pst[wave][pr][akb];
        *((v8bf*)&pa + 1) = *(const v8bf*)&Pst[wave][pr][akb + 8];

        // ---- O += P * V  (5 d-tiles of 16), batched frag loads ----
        const int vkb = (lane < 16) ? 0 : 16;
        v16bf bv16[5];
#pragma unroll
        for (int tt = 0; tt < 5; ++tt) {
            const int d = tt * 16 + (lane & 15);
            *(v8bf*)&bv16[tt]       = *(const v8bf*)&Vt[d][vkb];
            *((v8bf*)&bv16[tt] + 1) = *(const v8bf*)&Vt[d][vkb + 8];
        }
#pragma unroll
        for (int tt = 0; tt < 5; ++tt)
            acc[tt] = wmma_bf16(pa, bv16[tt], acc[tt]);
    }

    // ---- normalize and scatter to attn_out [B*SEQ][H_MODEL] (bf16) ----
    const int b = bh >> 4, h = bh & 15;
#pragma unroll
    for (int j = 0; j < 8; ++j) {
        const float inv = 1.0f / lrow[j];
        const int n = qrow0 + j + ((lane < 16) ? 0 : 8);
        if (n >= SEQ) continue;
#pragma unroll
        for (int tt = 0; tt < 5; ++tt) {
            const int d = tt * 16 + (lane & 15);
            if (d >= HEAD_DIM) continue;
            O[((size_t)(b * SEQ) + n) * H_MODEL + h * HEAD_DIM + d] =
                (__bf16)(acc[tt][j] * inv);
        }
    }
}

// ---------------------------------------------------------------------------
extern "C" void kernel_launch(void* const* d_in, const int* in_sizes, int n_in,
                              void* d_out, int out_size, void* d_ws, size_t ws_size,
                              hipStream_t stream) {
    const float* x  = (const float*)d_in[0];
    const float* wq = (const float*)d_in[1];
    const float* bq = (const float*)d_in[2];
    const float* wk = (const float*)d_in[3];
    const float* bk = (const float*)d_in[4];
    const float* wv = (const float*)d_in[5];
    const float* bv = (const float*)d_in[6];
    const float* wo = (const float*)d_in[7];
    const float* bo = (const float*)d_in[8];
    float* out = (float*)d_out;

    // workspace layout (bf16): Q | K | V (padded [B,H,736,96]) | attn_out
    __bf16* qbuf = (__bf16*)d_ws;
    __bf16* kbuf = qbuf + QKV_ELEMS;
    __bf16* vbuf = kbuf + QKV_ELEMS;
    __bf16* abuf = vbuf + QKV_ELEMS;   // [ROWS][H_MODEL]

    // 1) zero padded QKV (pad lanes must be exact zeros)
    const long long n16 = (long long)(3 * QKV_ELEMS) * 2 / 16;
    zero_ws<<<1024, 256, 0, stream>>>((uint4*)d_ws, n16);

    // 2) QKV projections (specialized kernels: no runtime mode dispatch)
    dim3 gp((ROWS + 127) / 128, H_MODEL / 128, 1);
    gemm_kern<0><<<gp, 256, 0, stream>>>(x, abuf, wq, bq, qbuf, out);
    gemm_kern<1><<<gp, 256, 0, stream>>>(x, abuf, wk, bk, kbuf, out);
    gemm_kern<2><<<gp, 256, 0, stream>>>(x, abuf, wv, bv, vbuf, out);

    // 3) flash attention per (b,h) x 64 query rows
    dim3 ga((SEQ + 63) / 64, BATCH * N_HEADS, 1);
    attn_wmma<<<ga, 128, 0, stream>>>(qbuf, kbuf, vbuf, abuf);

    // 4) output projection -> f32 d_out
    gemm_kern<3><<<gp, 256, 0, stream>>>(x, abuf, wo, bo, qbuf, out);
}
